// NoisyTopKGating_40235253629031
// MI455X (gfx1250) — matrix operand order repaired and checked by reading, not compiled, not run on hardware
//
#include <hip/hip_runtime.h>
#include <hip/hip_bf16.h>

typedef __attribute__((ext_vector_type(8)))  float  v8f;
typedef __attribute__((ext_vector_type(16))) __bf16 v16bf;
typedef __attribute__((ext_vector_type(8)))  __bf16 v8bf;

#define DD   960
#define HH1  128
#define HH2  32
#define EE   8
#define KC   64            // K-chunk for GEMM1
#define NCHUNK (DD / KC)   // 15
#define MT   128           // rows per block
#define LN_EPS 1e-5f

__device__ __forceinline__ v16bf load_afrag(const __bf16* p0, const __bf16* p1) {
    v8bf lo = *(const v8bf*)p0;   // K group +0  (16B)
    v8bf hi = *(const v8bf*)p1;   // K group +16 (16B)
    return __builtin_shufflevector(lo, hi, 0,1,2,3,4,5,6,7,8,9,10,11,12,13,14,15);
}

__device__ __forceinline__ float gelu_exact(float v) {
    return 0.5f * v * (1.0f + erff(v * 0.70710678118654752440f));
}

__global__ __launch_bounds__(256)
void gate_fused_kernel(const float* __restrict__ x,
                       const float* __restrict__ W1, const float* __restrict__ b1,
                       const float* __restrict__ g1, const float* __restrict__ be1,
                       const float* __restrict__ W2, const float* __restrict__ b2,
                       const float* __restrict__ g2, const float* __restrict__ be2,
                       const float* __restrict__ W3, const float* __restrict__ b3,
                       float* __restrict__ outW, float* __restrict__ outI,
                       float* __restrict__ outL)
{
    // 2 x 32KB staging buffers: [A tile 128x64 bf16][B tile 128x64 bf16]
    // buffer 0 region is later reused as h1 (128x128 bf16 = 32KB)
    __shared__ alignas(32) __bf16 smemAB[2 * (MT * KC + HH1 * KC)];
    __shared__ alignas(32) __bf16 smemW2[HH2 * HH1];   // 8KB transposed W2 [n][k]
    __shared__ alignas(16) float  smemH2[MT * HH2];    // 16KB
    __shared__ float smemW3[HH2 * EE];
    __shared__ float smemB3[EE];

    __bf16* ldsH1 = smemAB;            // [MT][HH1]  (aliases buffer 0)

    const int tid   = threadIdx.x;
    const int wave  = tid >> 5;
    const int lane  = tid & 31;
    const int laneM = lane & 15;
    const int laneH = lane >> 4;
    const int mBase = wave * 16;
    const int blockRow = blockIdx.x * MT;

    // per-thread staging coordinates (constant across chunks)
    const int aRow = (tid * 8) >> 4 >= 0 ? 0 : 0; // (placeholder to keep structure simple)
    // A tile: idx4 = tid + i*256 -> row = idx4>>4, kq = (idx4&15)*4
    // B tile: idx4 = tid + i*256 -> kk  = idx4>>5, n4 = (idx4&31)*4

    // ---- stage W2 transposed + W3 + b3 (once) ----
    #pragma unroll
    for (int i = 0; i < 16; ++i) {
        int idx = tid + i * 256;       // HH1*HH2 = 4096 elements
        int k = idx >> 5;              // W2 row-major [k][n]
        int n = idx & 31;
        smemW2[n * HH1 + k] = (__bf16)W2[idx];
    }
    if (tid < HH2 * EE) smemW3[tid] = W3[tid];
    if (tid < EE)       smemB3[tid] = b3[tid];

    // per-lane layer-1 params for column c = nt*16 + laneM
    float b1r[8], g1r[8], be1r[8];
    #pragma unroll
    for (int nt = 0; nt < 8; ++nt) {
        int c = nt * 16 + laneM;
        b1r[nt] = b1[c]; g1r[nt] = g1[c]; be1r[nt] = be1[c];
    }
    float b2r[2], g2r[2], be2r[2];
    #pragma unroll
    for (int nt = 0; nt < 2; ++nt) {
        int c = nt * 16 + laneM;
        b2r[nt] = b2[c]; g2r[nt] = g2[c]; be2r[nt] = be2[c];
    }

    // ======================= GEMM1: x @ W1 (software-pipelined) =======================
    v8f acc[8] = {};
    float4 ax[8], bw[8];               // in-flight global data for the next chunk

    // prologue: load + stage chunk 0 into buffer 0
    {
        const int k0 = 0;
        #pragma unroll
        for (int i = 0; i < 8; ++i) {
            int idx4 = tid + i * 256;
            int row  = idx4 >> 4;
            int kq   = (idx4 & 15) * 4;
            ax[i] = *(const float4*)(x + (size_t)(blockRow + row) * DD + k0 + kq);
            int kk   = idx4 >> 5;
            int n4   = (idx4 & 31) * 4;
            bw[i] = *(const float4*)(W1 + (size_t)(k0 + kk) * HH1 + n4);
        }
        __bf16* ldsA = smemAB;
        __bf16* ldsB = smemAB + MT * KC;
        #pragma unroll
        for (int i = 0; i < 8; ++i) {
            int idx4 = tid + i * 256;
            int row  = idx4 >> 4;
            int kq   = (idx4 & 15) * 4;
            __bf16* p = ldsA + row * KC + kq;
            p[0] = (__bf16)ax[i].x; p[1] = (__bf16)ax[i].y;
            p[2] = (__bf16)ax[i].z; p[3] = (__bf16)ax[i].w;
            int kk   = idx4 >> 5;
            int n4   = (idx4 & 31) * 4;
            ldsB[(n4 + 0) * KC + kk] = (__bf16)bw[i].x;
            ldsB[(n4 + 1) * KC + kk] = (__bf16)bw[i].y;
            ldsB[(n4 + 2) * KC + kk] = (__bf16)bw[i].z;
            ldsB[(n4 + 3) * KC + kk] = (__bf16)bw[i].w;
        }
    }

    for (int ch = 0; ch < NCHUNK; ++ch) {
        __syncthreads();   // buf[ch&1] fully staged; buf[(ch+1)&1] free to overwrite

        // issue global loads for chunk ch+1 (latency hidden behind the WMMAs below)
        const bool more = (ch + 1 < NCHUNK);
        if (more) {
            const int k1 = (ch + 1) * KC;
            #pragma unroll
            for (int i = 0; i < 8; ++i) {
                int idx4 = tid + i * 256;
                int row  = idx4 >> 4;
                int kq   = (idx4 & 15) * 4;
                ax[i] = *(const float4*)(x + (size_t)(blockRow + row) * DD + k1 + kq);
                int kk   = idx4 >> 5;
                int n4   = (idx4 & 31) * 4;
                bw[i] = *(const float4*)(W1 + (size_t)(k1 + kk) * HH1 + n4);
                if (ch + 2 < NCHUNK)   // global_prefetch_b8, distance 2
                    __builtin_prefetch(x + (size_t)(blockRow + row) * DD + k1 + KC + kq, 0, 0);
            }
        }

        // compute chunk ch from buf[ch&1]: 2 k-steps x 8 n-tiles
        {
            const __bf16* ldsA = smemAB + (ch & 1) * (MT * KC + HH1 * KC);
            const __bf16* ldsB = ldsA + MT * KC;
            #pragma unroll
            for (int ks = 0; ks < 2; ++ks) {
                const int abase = ks * 32 + laneH * 8;    // A: groups {+0,+16}
                const int bbase = ks * 32 + laneH * 8;    // B: groups {+0,+16}
                const int arow  = mBase + laneM;
                v16bf afrag = load_afrag(ldsA + arow * KC + abase,
                                         ldsA + arow * KC + abase + 16);
                #pragma unroll
                for (int nt = 0; nt < 8; ++nt) {
                    int n = nt * 16 + laneM;
                    v16bf bfrag = load_afrag(ldsB + n * KC + bbase, ldsB + n * KC + bbase + 16);
                    acc[nt] = __builtin_amdgcn_wmma_f32_16x16x32_bf16(
                        false, afrag, false, bfrag, (short)0, acc[nt], false, false);
                }
            }
        }

        // convert + store chunk ch+1 into the other buffer
        if (more) {
            __bf16* ldsA = smemAB + ((ch + 1) & 1) * (MT * KC + HH1 * KC);
            __bf16* ldsB = ldsA + MT * KC;
            #pragma unroll
            for (int i = 0; i < 8; ++i) {
                int idx4 = tid + i * 256;
                int row  = idx4 >> 4;
                int kq   = (idx4 & 15) * 4;
                __bf16* p = ldsA + row * KC + kq;
                p[0] = (__bf16)ax[i].x; p[1] = (__bf16)ax[i].y;
                p[2] = (__bf16)ax[i].z; p[3] = (__bf16)ax[i].w;
                int kk   = idx4 >> 5;
                int n4   = (idx4 & 31) * 4;
                ldsB[(n4 + 0) * KC + kk] = (__bf16)bw[i].x;
                ldsB[(n4 + 1) * KC + kk] = (__bf16)bw[i].y;
                ldsB[(n4 + 2) * KC + kk] = (__bf16)bw[i].z;
                ldsB[(n4 + 3) * KC + kk] = (__bf16)bw[i].w;
            }
        }
    }

    // ---- bias + LayerNorm(H1=128) + exact GELU, in WMMA C-layout ----
    #pragma unroll
    for (int r = 0; r < 8; ++r) {
        float s = 0.f, ss = 0.f;
        #pragma unroll
        for (int nt = 0; nt < 8; ++nt) {
            float v = acc[nt][r] + b1r[nt];
            acc[nt][r] = v;
            s += v; ss += v * v;
        }
        #pragma unroll
        for (int m = 1; m <= 8; m <<= 1) {       // reduce across the 16-lane half
            s  += __shfl_xor(s,  m, 32);
            ss += __shfl_xor(ss, m, 32);
        }
        float mean = s * (1.0f / HH1);
        float var  = ss * (1.0f / HH1) - mean * mean;
        float rstd = rsqrtf(var + LN_EPS);
        #pragma unroll
        for (int nt = 0; nt < 8; ++nt) {
            float v = (acc[nt][r] - mean) * rstd * g1r[nt] + be1r[nt];
            acc[nt][r] = gelu_exact(v);
        }
    }

    __syncthreads();   // all waves finished reading staging buffers (h1 aliases buffer 0)
    #pragma unroll
    for (int r = 0; r < 8; ++r) {
        int row = mBase + r + 8 * laneH;         // C-layout row
        #pragma unroll
        for (int nt = 0; nt < 8; ++nt)
            ldsH1[row * HH1 + nt * 16 + laneM] = (__bf16)acc[nt][r];
    }
    __syncthreads();

    // ======================= GEMM2: h1 @ W2 =======================
    v8f acc2[2] = {};
    #pragma unroll
    for (int ks = 0; ks < 4; ++ks) {
        const int abase = ks * 32 + laneH * 8;
        const int bbase = ks * 32 + laneH * 8;
        const int arow  = mBase + laneM;
        v16bf afrag = load_afrag(ldsH1 + arow * HH1 + abase,
                                 ldsH1 + arow * HH1 + abase + 16);
        #pragma unroll
        for (int nt = 0; nt < 2; ++nt) {
            int n = nt * 16 + laneM;
            v16bf bfrag = load_afrag(smemW2 + n * HH1 + bbase, smemW2 + n * HH1 + bbase + 16);
            acc2[nt] = __builtin_amdgcn_wmma_f32_16x16x32_bf16(
                false, afrag, false, bfrag, (short)0, acc2[nt], false, false);
        }
    }

    // ---- bias + LayerNorm(H2=32) + GELU ----
    #pragma unroll
    for (int r = 0; r < 8; ++r) {
        float v0 = acc2[0][r] + b2r[0];
        float v1 = acc2[1][r] + b2r[1];
        float s = v0 + v1, ss = v0 * v0 + v1 * v1;
        #pragma unroll
        for (int m = 1; m <= 8; m <<= 1) {
            s  += __shfl_xor(s,  m, 32);
            ss += __shfl_xor(ss, m, 32);
        }
        float mean = s * (1.0f / HH2);
        float var  = ss * (1.0f / HH2) - mean * mean;
        float rstd = rsqrtf(var + LN_EPS);
        acc2[0][r] = gelu_exact((v0 - mean) * rstd * g2r[0] + be2r[0]);
        acc2[1][r] = gelu_exact((v1 - mean) * rstd * g2r[1] + be2r[1]);
    }

    __syncthreads();
    #pragma unroll
    for (int r = 0; r < 8; ++r) {
        int row = mBase + r + 8 * laneH;
        smemH2[row * HH2 + laneM]      = acc2[0][r];
        smemH2[row * HH2 + 16 + laneM] = acc2[1][r];
    }
    __syncthreads();

    // ============ GEMM3 (32x8) + top-3 + softmax: one thread per row ============
    if (tid < MT) {
        const int gRow = blockRow + tid;
        float lg[EE];
        #pragma unroll
        for (int e = 0; e < EE; ++e) lg[e] = smemB3[e];
        #pragma unroll
        for (int c = 0; c < HH2; ++c) {
            float h = smemH2[tid * HH2 + c];
            #pragma unroll
            for (int e = 0; e < EE; ++e) lg[e] += h * smemW3[c * EE + e];
        }
        float tmp[EE];
        #pragma unroll
        for (int e = 0; e < EE; ++e) tmp[e] = lg[e];
        float wv[3]; int wi[3];
        #pragma unroll
        for (int t = 0; t < 3; ++t) {            // strict '>' => first index wins ties (matches lax.top_k)
            float best = tmp[0]; int bi = 0;
            #pragma unroll
            for (int e = 1; e < EE; ++e) if (tmp[e] > best) { best = tmp[e]; bi = e; }
            wv[t] = best; wi[t] = bi; tmp[bi] = -3.0e38f;
        }
        float m0 = wv[0];
        float e0 = __expf(wv[0] - m0), e1 = __expf(wv[1] - m0), e2 = __expf(wv[2] - m0);
        float inv = 1.0f / (e0 + e1 + e2);
        outW[(size_t)gRow * 3 + 0] = e0 * inv;
        outW[(size_t)gRow * 3 + 1] = e1 * inv;
        outW[(size_t)gRow * 3 + 2] = e2 * inv;
        outI[(size_t)gRow * 3 + 0] = (float)wi[0];
        outI[(size_t)gRow * 3 + 1] = (float)wi[1];
        outI[(size_t)gRow * 3 + 2] = (float)wi[2];
        #pragma unroll
        for (int e = 0; e < EE; ++e) outL[(size_t)gRow * EE + e] = lg[e];
    }
}

extern "C" void kernel_launch(void* const* d_in, const int* in_sizes, int n_in,
                              void* d_out, int out_size, void* d_ws, size_t ws_size,
                              hipStream_t stream) {
    const float* x   = (const float*)d_in[0];
    const float* W1  = (const float*)d_in[1];
    const float* b1  = (const float*)d_in[2];
    const float* g1  = (const float*)d_in[3];
    const float* be1 = (const float*)d_in[4];
    const float* W2  = (const float*)d_in[5];
    const float* b2  = (const float*)d_in[6];
    const float* g2  = (const float*)d_in[7];
    const float* be2 = (const float*)d_in[8];
    const float* W3  = (const float*)d_in[9];
    const float* b3  = (const float*)d_in[10];

    const int Btot = in_sizes[0] / DD;           // 131072
    float* out  = (float*)d_out;                 // [weights B*3 | indices B*3 | logits B*8]
    float* outW = out;
    float* outI = out + (size_t)Btot * 3;
    float* outL = out + (size_t)Btot * 6;

    dim3 grid(Btot / MT), block(256);
    gate_fused_kernel<<<grid, block, 0, stream>>>(x, W1, b1, g1, be1,
                                                  W2, b2, g2, be2, W3, b3,
                                                  outW, outI, outL);
}
